// CrossAttentionBlock_29291676959395
// MI455X (gfx1250) — compile-verified, matrix operand order checked by reading, not run on hardware
//
#include <hip/hip_runtime.h>

// ---------------------------------------------------------------------------
// CDNA5 / gfx1250 cross-attention block. wave32, WMMA f32_16x16x32_f16.
// Problem constants from the reference.
// ---------------------------------------------------------------------------
#define B_   2
#define L_   2048
#define LC_  1024
#define DM_  1024
#define DH_  64
#define NH_  16

typedef _Float16 v16h __attribute__((ext_vector_type(16)));
typedef _Float16 v8h  __attribute__((ext_vector_type(8)));
typedef _Float16 v4h  __attribute__((ext_vector_type(4)));
typedef float    v8f  __attribute__((ext_vector_type(8)));
typedef float    f32x4 __attribute__((ext_vector_type(4)));

union F16Frag { v16h v; v8h h[2]; };

static __device__ __forceinline__ v8f zero8() {
  v8f z;
#pragma unroll
  for (int i = 0; i < 8; ++i) z[i] = 0.0f;
  return z;
}

static __device__ __forceinline__ float redsum32(float v) {
#pragma unroll
  for (int m = 1; m < 32; m <<= 1) v += __shfl_xor(v, m, 32);
  return v;
}

// ---------------------------------------------------------------------------
// Weight cast + transpose: w[K][N] f32  ->  wT[N][K] f16
// (so every WMMA B-fragment is two contiguous 16-byte loads along K)
// ---------------------------------------------------------------------------
__global__ __launch_bounds__(256) void cast_transpose_k(
    const float* __restrict__ w, _Float16* __restrict__ wT, int K, int N) {
  size_t tid = (size_t)blockIdx.x * 256 + threadIdx.x;
  if (tid >= (size_t)K * N) return;
  int n = (int)(tid / K);
  int k = (int)(tid % K);
  wT[tid] = (_Float16)w[(size_t)k * N + n];
}

// ---------------------------------------------------------------------------
// RMS norm a row of DM_ f32, scale, cast to f16. One 256-thread block per row.
// ---------------------------------------------------------------------------
__global__ __launch_bounds__(256) void rmsnorm_cast_k(
    const float* __restrict__ x, const float* __restrict__ g,
    _Float16* __restrict__ out) {
  __shared__ float red[8];
  const int tid = threadIdx.x;
  const size_t row = blockIdx.x;
  const float* xr = x + row * DM_;
  f32x4 v = ((const f32x4*)xr)[tid];
  float ss = v[0]*v[0] + v[1]*v[1] + v[2]*v[2] + v[3]*v[3];
  ss = redsum32(ss);
  if ((tid & 31) == 0) red[tid >> 5] = ss;
  __syncthreads();
  float tot = 0.f;
#pragma unroll
  for (int i = 0; i < 8; ++i) tot += red[i];
  float r = rsqrtf(tot / (float)DM_ + 1e-6f);
  f32x4 gg = ((const f32x4*)g)[tid];
  v4h o;
  o[0] = (_Float16)(v[0] * gg[0] * r);
  o[1] = (_Float16)(v[1] * gg[1] * r);
  o[2] = (_Float16)(v[2] * gg[2] * r);
  o[3] = (_Float16)(v[3] * gg[3] * r);
  ((v4h*)(out + row * DM_))[tid] = o;
}

// ---------------------------------------------------------------------------
// WMMA GEMM: C[M][N] f32 = A[M][K] f16 @ BT[N][K]^T f16 (+ optional f32 res)
// 8 waves / block, each wave owns a 32x64 output tile (2x4 accumulators):
// two A fragments share four B fragments -> 8 WMMAs per 12x16B load batch.
// A/B fragments per ISA 7.12.2: lane = row (l&15), two 16B loads per frag.
// K-tile two ahead (k0+64) prefetched with global_prefetch_b8 (speculative, tail-safe).
// ---------------------------------------------------------------------------
__global__ __launch_bounds__(256) void gemm_wmma_k(
    const _Float16* __restrict__ A, const _Float16* __restrict__ BT,
    const float* __restrict__ res, float* __restrict__ C,
    int M, int N, int K) {
  const int lane = threadIdx.x & 31;
  const int wave = blockIdx.x * 8 + (threadIdx.x >> 5);
  const int nt64 = N >> 6;
  const int mt = wave / nt64;            // 32-row tile index
  const int nt = wave % nt64;            // 64-col tile index
  const int mloc = lane & 15;
  const int half = lane >> 4;
  const int koff = half << 3;
  const _Float16* arow0 = A + (size_t)(mt * 32 + mloc) * K;
  const _Float16* arow1 = arow0 + (size_t)16 * K;
  const _Float16* bbase = BT + (size_t)(nt * 64 + mloc) * K;

  v8f acc[2][4];
#pragma unroll
  for (int u = 0; u < 2; ++u)
#pragma unroll
    for (int t = 0; t < 4; ++t) acc[u][t] = zero8();

  for (int k0 = 0; k0 < K; k0 += 32) {
    F16Frag a0, a1;
    a0.h[0] = *(const v8h*)(arow0 + k0 + koff);
    a0.h[1] = *(const v8h*)(arow0 + k0 + 16 + koff);
    a1.h[0] = *(const v8h*)(arow1 + k0 + koff);
    a1.h[1] = *(const v8h*)(arow1 + k0 + 16 + koff);
    __builtin_prefetch(arow0 + k0 + 64, 0, 1);   // k-tile two ahead, row block 0
    __builtin_prefetch(arow1 + k0 + 64, 0, 1);   // k-tile two ahead, row block 1
#pragma unroll
    for (int t = 0; t < 4; ++t) {
      const _Float16* brow = bbase + (size_t)t * 16 * K;
      F16Frag b;
      b.h[0] = *(const v8h*)(brow + k0 + koff);
      b.h[1] = *(const v8h*)(brow + k0 + 16 + koff);
      __builtin_prefetch(brow + k0 + 64, 0, 1);  // k-tile two ahead of this B row
      acc[0][t] = __builtin_amdgcn_wmma_f32_16x16x32_f16(
          false, a0.v, false, b.v, (short)0, acc[0][t], false, false);
      acc[1][t] = __builtin_amdgcn_wmma_f32_16x16x32_f16(
          false, a1.v, false, b.v, (short)0, acc[1][t], false, false);
    }
  }

#pragma unroll
  for (int u = 0; u < 2; ++u) {
#pragma unroll
    for (int t = 0; t < 4; ++t) {
#pragma unroll
      for (int r = 0; r < 8; ++r) {
        size_t idx = (size_t)(mt * 32 + u * 16 + r + 8 * half) * N
                   + nt * 64 + t * 16 + mloc;
        float v = acc[u][t][r];
        if (res) v += res[idx];
        C[idx] = v;
      }
    }
  }
}

// ---------------------------------------------------------------------------
// Q epilogue: cosine-normalize each 64-dim head vector, apply 2D RoPE,
// write q_h in [B,H,L,64] f16. One wave per (b,l,h); lane holds e, e+32.
// ---------------------------------------------------------------------------
__global__ __launch_bounds__(256) void q_epilogue_k(
    const float* __restrict__ qf, const float* __restrict__ pos,
    const float* __restrict__ scale, const float* __restrict__ freqs,
    _Float16* __restrict__ qh) {
  int gw = blockIdx.x * 8 + (threadIdx.x >> 5);
  int lane = threadIdx.x & 31;
  int l  = gw % L_;
  int bh = gw / L_;
  int b = bh >> 4, h = bh & 15;
  const float* row = qf + (size_t)(b * L_ + l) * DM_;
  float q1 = row[h * DH_ + lane];
  float q2 = row[h * DH_ + lane + 32];
  float ss = redsum32(q1 * q1 + q2 * q2);
  float rn = sqrtf(scale[h]) * rsqrtf(ss + 1e-6f);
  q1 *= rn; q2 *= rn;
  // theta[t] = (t<16 ? pos0 : pos1) * freqs[h][t%16], t = lane in 0..31
  float p = pos[(size_t)(b * L_ + l) * 2 + (lane < 16 ? 0 : 1)];
  float f = freqs[h * 16 + (lane & 15)];
  float th = p * f;
  float s, c;
  __sincosf(th, &s, &c);
  size_t qb = ((size_t)bh * L_ + l) * DH_;
  qh[qb + lane]      = (_Float16)(q1 * c - q2 * s);
  qh[qb + lane + 32] = (_Float16)(q2 * c + q1 * s);
}

// ---------------------------------------------------------------------------
// KV epilogue: k cosine-normalized -> k_h [B,H,LC,64]; v -> vT [B,H,64,LC]
// (v transposed so P@V B-fragments are contiguous along keys).
// ---------------------------------------------------------------------------
__global__ __launch_bounds__(256) void kv_epilogue_k(
    const float* __restrict__ kvf, const float* __restrict__ scale,
    _Float16* __restrict__ kh, _Float16* __restrict__ vT) {
  int gw = blockIdx.x * 8 + (threadIdx.x >> 5);
  int lane = threadIdx.x & 31;
  int lc = gw % LC_;
  int bh = gw / LC_;
  int b = bh >> 4, h = bh & 15;
  const float* row = kvf + (size_t)(b * LC_ + lc) * (2 * DM_);
  float k1 = row[h * DH_ + lane];
  float k2 = row[h * DH_ + lane + 32];
  float ss = redsum32(k1 * k1 + k2 * k2);
  float rn = sqrtf(scale[h]) * rsqrtf(ss + 1e-6f);
  size_t kb = ((size_t)bh * LC_ + lc) * DH_;
  kh[kb + lane]      = (_Float16)(k1 * rn);
  kh[kb + lane + 32] = (_Float16)(k2 * rn);
  float v1 = row[DM_ + h * DH_ + lane];
  float v2 = row[DM_ + h * DH_ + lane + 32];
  vT[((size_t)bh * DH_ + lane)      * LC_ + lc] = (_Float16)v1;
  vT[((size_t)bh * DH_ + lane + 32) * LC_ + lc] = (_Float16)v2;
}

// ---------------------------------------------------------------------------
// Flash attention, one pass over keys. Block = 4 waves; wave = 16 query rows.
// Per 32-key tile: 2x2 WMMAs for S, online softmax in C-layout (xor-shuffle
// row reductions within lane halves), P staged via LDS into A-fragment
// layout, 4 WMMAs for O += P@V. Output o_h [B, L, D_MODEL] f16.
// ---------------------------------------------------------------------------
__global__ __launch_bounds__(128) void attn_wmma_k(
    const _Float16* __restrict__ qh, const _Float16* __restrict__ kh,
    const _Float16* __restrict__ vT, _Float16* __restrict__ oh) {
  __shared__ __align__(16) _Float16 Plds[4][16 * 32];
  const int lane = threadIdx.x & 31;
  const int w    = threadIdx.x >> 5;
  const int nqb  = L_ / 64;
  const int bh   = blockIdx.x / nqb;
  const int qb   = blockIdx.x % nqb;
  const int b = bh >> 4, h = bh & 15;
  const int m0   = qb * 64 + w * 16;
  const int mloc = lane & 15;
  const int half = lane >> 4;
  const int koff = half << 3;

  const _Float16* qrow = qh + ((size_t)bh * L_ + m0 + mloc) * DH_;
  F16Frag aq0, aq1;                       // q rows, e=0..31 and e=32..63
  aq0.h[0] = *(const v8h*)(qrow + koff);
  aq0.h[1] = *(const v8h*)(qrow + 16 + koff);
  aq1.h[0] = *(const v8h*)(qrow + 32 + koff);
  aq1.h[1] = *(const v8h*)(qrow + 48 + koff);

  v8f O[4];
#pragma unroll
  for (int t = 0; t < 4; ++t) O[t] = zero8();
  float mrow[8], lrow[8];
#pragma unroll
  for (int r = 0; r < 8; ++r) { mrow[r] = -3.0e38f; lrow[r] = 0.f; }

  const _Float16* kbase = kh + (size_t)bh * LC_ * DH_;
  const _Float16* vbase = vT + (size_t)bh * DH_ * LC_;
  _Float16* pl = &Plds[w][0];

  for (int j = 0; j < LC_; j += 32) {
    v8f sL = zero8(), sR = zero8();
    {
      const _Float16* kr = kbase + (size_t)(j + mloc) * DH_;
      F16Frag b0, b1;
      b0.h[0] = *(const v8h*)(kr + koff);
      b0.h[1] = *(const v8h*)(kr + 16 + koff);
      b1.h[0] = *(const v8h*)(kr + 32 + koff);
      b1.h[1] = *(const v8h*)(kr + 48 + koff);
      sL = __builtin_amdgcn_wmma_f32_16x16x32_f16(false, aq0.v, false, b0.v, (short)0, sL, false, false);
      sL = __builtin_amdgcn_wmma_f32_16x16x32_f16(false, aq1.v, false, b1.v, (short)0, sL, false, false);
    }
    {
      const _Float16* kr = kbase + (size_t)(j + 16 + mloc) * DH_;
      F16Frag b0, b1;
      b0.h[0] = *(const v8h*)(kr + koff);
      b0.h[1] = *(const v8h*)(kr + 16 + koff);
      b1.h[0] = *(const v8h*)(kr + 32 + koff);
      b1.h[1] = *(const v8h*)(kr + 48 + koff);
      sR = __builtin_amdgcn_wmma_f32_16x16x32_f16(false, aq0.v, false, b0.v, (short)0, sR, false, false);
      sR = __builtin_amdgcn_wmma_f32_16x16x32_f16(false, aq1.v, false, b1.v, (short)0, sR, false, false);
    }

    // --- online softmax in C-layout (row r+8*half, col = lane&15) ---
    float pmax[8];
#pragma unroll
    for (int r = 0; r < 8; ++r) pmax[r] = fmaxf(sL[r], sR[r]);
#pragma unroll
    for (int m = 1; m < 16; m <<= 1) {
#pragma unroll
      for (int r = 0; r < 8; ++r)
        pmax[r] = fmaxf(pmax[r], __shfl_xor(pmax[r], m, 32));
    }
    float corr[8], rs[8];
    v8f pLv, pRv;
#pragma unroll
    for (int r = 0; r < 8; ++r) {
      float mn = fmaxf(mrow[r], pmax[r]);
      corr[r] = __expf(mrow[r] - mn);
      mrow[r] = mn;
      pLv[r] = __expf(sL[r] - mn);
      pRv[r] = __expf(sR[r] - mn);
      rs[r] = pLv[r] + pRv[r];
    }
#pragma unroll
    for (int m = 1; m < 16; m <<= 1) {
#pragma unroll
      for (int r = 0; r < 8; ++r) rs[r] += __shfl_xor(rs[r], m, 32);
    }
#pragma unroll
    for (int r = 0; r < 8; ++r) {
      lrow[r] = lrow[r] * corr[r] + rs[r];
      O[0][r] *= corr[r]; O[1][r] *= corr[r];
      O[2][r] *= corr[r]; O[3][r] *= corr[r];
    }

    // --- stage P (16x32 f16) in LDS, reload in A-fragment layout ---
#pragma unroll
    for (int r = 0; r < 8; ++r) {
      int prow = r + 8 * half;
      pl[prow * 32 + mloc]      = (_Float16)pLv[r];
      pl[prow * 32 + 16 + mloc] = (_Float16)pRv[r];
    }
    F16Frag ap;   // same-wave LDS RAW: in-order DS, compiler waits dscnt
    ap.h[0] = *(const v8h*)(pl + mloc * 32 + koff);
    ap.h[1] = *(const v8h*)(pl + mloc * 32 + 16 + koff);

    // --- O += P @ V (v stored [e][key], contiguous along keys) ---
#pragma unroll
    for (int t = 0; t < 4; ++t) {
      const _Float16* vr = vbase + (size_t)(t * 16 + mloc) * LC_ + j;
      F16Frag bv;
      bv.h[0] = *(const v8h*)(vr + koff);
      bv.h[1] = *(const v8h*)(vr + 16 + koff);
      O[t] = __builtin_amdgcn_wmma_f32_16x16x32_f16(
          false, ap.v, false, bv.v, (short)0, O[t], false, false);
    }
  }

#pragma unroll
  for (int r = 0; r < 8; ++r) {
    float inv = 1.0f / lrow[r];
    int row = m0 + r + 8 * half;
    size_t base = (size_t)(b * L_ + row) * DM_ + h * DH_;
#pragma unroll
    for (int t = 0; t < 4; ++t)
      oh[base + t * 16 + mloc] = (_Float16)(O[t][r] * inv);
  }
}

// ---------------------------------------------------------------------------
// Host launcher
// ---------------------------------------------------------------------------
extern "C" void kernel_launch(void* const* d_in, const int* in_sizes, int n_in,
                              void* d_out, int out_size, void* d_ws, size_t ws_size,
                              hipStream_t stream) {
  (void)in_sizes; (void)n_in; (void)out_size; (void)ws_size;
  const float* x    = (const float*)d_in[0];
  const float* pos  = (const float*)d_in[1];
  const float* xc   = (const float*)d_in[2];
  const float* nsc  = (const float*)d_in[3];
  const float* ncsc = (const float*)d_in[4];
  const float* qw   = (const float*)d_in[5];
  const float* kvw  = (const float*)d_in[6];
  const float* scl  = (const float*)d_in[7];
  const float* outw = (const float*)d_in[8];
  const float* frq  = (const float*)d_in[9];
  float* out = (float*)d_out;

  char* ws = (char*)d_ws;
  size_t off = 0;
  auto alloc = [&](size_t bytes) -> void* {
    void* p = ws + off;
    off += (bytes + 255) & ~(size_t)255;
    return p;
  };
  _Float16* qwT  = (_Float16*)alloc((size_t)DM_ * DM_ * 2);
  _Float16* kvwT = (_Float16*)alloc((size_t)2 * DM_ * DM_ * 2);
  _Float16* owT  = (_Float16*)alloc((size_t)DM_ * DM_ * 2);
  _Float16* xn   = (_Float16*)alloc((size_t)B_ * L_ * DM_ * 2);
  _Float16* xcn  = (_Float16*)alloc((size_t)B_ * LC_ * DM_ * 2);
  float*    qf   = (float*)   alloc((size_t)B_ * L_ * DM_ * 4);
  float*    kvf  = (float*)   alloc((size_t)B_ * LC_ * 2 * DM_ * 4);
  _Float16* qh   = (_Float16*)alloc((size_t)B_ * NH_ * L_ * DH_ * 2);
  _Float16* kh   = (_Float16*)alloc((size_t)B_ * NH_ * LC_ * DH_ * 2);
  _Float16* vT   = (_Float16*)alloc((size_t)B_ * NH_ * DH_ * LC_ * 2);
  _Float16* oh   = (_Float16*)alloc((size_t)B_ * L_ * DM_ * 2);

  // weights -> f16, transposed to [N][K]
  cast_transpose_k<<<(DM_ * DM_) / 256, 256, 0, stream>>>(qw,  qwT,  DM_, DM_);
  cast_transpose_k<<<(2 * DM_ * DM_) / 256, 256, 0, stream>>>(kvw, kvwT, DM_, 2 * DM_);
  cast_transpose_k<<<(DM_ * DM_) / 256, 256, 0, stream>>>(outw, owT, DM_, DM_);

  // RMS norms
  rmsnorm_cast_k<<<B_ * L_,  256, 0, stream>>>(x,  nsc,  xn);
  rmsnorm_cast_k<<<B_ * LC_, 256, 0, stream>>>(xc, ncsc, xcn);

  // projections (32-row tiles per wave)
  gemm_wmma_k<<<(B_ * L_ / 32) * (DM_ / 64) / 8, 256, 0, stream>>>(
      xn, qwT, nullptr, qf, B_ * L_, DM_, DM_);
  gemm_wmma_k<<<(B_ * LC_ / 32) * (2 * DM_ / 64) / 8, 256, 0, stream>>>(
      xcn, kvwT, nullptr, kvf, B_ * LC_, 2 * DM_, DM_);

  // epilogues: cos-norm + RoPE on q; cos-norm k, transpose v
  q_epilogue_k<<<B_ * NH_ * L_ / 8, 256, 0, stream>>>(qf, pos, scl, frq, qh);
  kv_epilogue_k<<<B_ * NH_ * LC_ / 8, 256, 0, stream>>>(kvf, scl, kh, vT);

  // flash attention
  attn_wmma_k<<<B_ * NH_ * (L_ / 64), 128, 0, stream>>>(qh, kh, vT, oh);

  // output projection + residual, f32 out
  gemm_wmma_k<<<(B_ * L_ / 32) * (DM_ / 64) / 8, 256, 0, stream>>>(
      oh, owT, x, out, B_ * L_, DM_, DM_);
}
